// Kmeans_44547400794407
// MI455X (gfx1250) — matrix operand order repaired and checked by reading, not compiled, not run on hardware
//
#include <hip/hip_runtime.h>

typedef __attribute__((ext_vector_type(16))) __bf16 v16bf;
typedef __attribute__((ext_vector_type(8)))  float  v8f;
typedef __attribute__((ext_vector_type(4)))  float  v4f;

#define NPTS 131072
#define DIM  256
#define KC   64
#define BLK_PTS 256
#define NBLOCKS (NPTS / BLK_PTS)   // 512 blocks
#define NSTRIPS (BLK_PTS / 128)    // 2 strips of 128 rows (8 waves x 16)
#define ITERS 20

// ---------------------------------------------------------------------------
// init: protos = x[:64] (f32 + bf16 mirror); p_sq; zero sums/counts/done.
// grid = 64 blocks x 256 threads
// ---------------------------------------------------------------------------
__global__ void kmeans_init(const float* __restrict__ x,
                            float* __restrict__ protos, __bf16* __restrict__ pbf,
                            float* __restrict__ psq,
                            float* __restrict__ sums, float* __restrict__ counts,
                            int* __restrict__ done) {
    __shared__ float red[256];
    const int b = blockIdx.x, tid = threadIdx.x;
    const float v = x[b * DIM + tid];
    protos[b * DIM + tid] = v;
    pbf[b * DIM + tid]    = (__bf16)v;
    sums[b * DIM + tid]   = 0.0f;
    red[tid] = v * v;
    __syncthreads();
    for (int s = 128; s > 0; s >>= 1) {
        if (tid < s) red[tid] += red[tid + s];
        __syncthreads();
    }
    if (tid == 0) {
        psq[b] = red[0];
        counts[b] = 0.0f;
        if (b == 0) *done = 0;
    }
}

// ---------------------------------------------------------------------------
// assign: WMMA bf16 distance GEMM + argmin + LDS segment-sum accumulation
// grid = 512 blocks x 256 threads (8 waves); each block handles 256 points.
// ---------------------------------------------------------------------------
__global__ void __launch_bounds__(256)
kmeans_assign(const float* __restrict__ x, const __bf16* __restrict__ pbf,
              const float* __restrict__ psq, int* __restrict__ idx,
              float* __restrict__ sums, float* __restrict__ counts,
              const int* __restrict__ done) {
    if (*done) return;  // uniform: converged -> keep previous idx/protos

    __shared__ float lsum[KC * DIM];       // 64 KB block-local centroid sums
    __shared__ float lcnt[KC];
    __shared__ float d2s[8][16 * 64];      // per-wave 16x64 distance tile

    const int tid  = threadIdx.x;
    const int wave = tid >> 5;
    const int lane = tid & 31;
    const int hi   = lane >> 4;            // lane-half selector
    const int lo   = lane & 15;

    for (int i = tid; i < KC * DIM; i += 256) lsum[i] = 0.0f;
    if (tid < KC) lcnt[tid] = 0.0f;
    __syncthreads();

    // p_sq for this lane's 4 cluster columns (n = t*16 + lo)
    float psq_l[4];
#pragma unroll
    for (int t = 0; t < 4; ++t) psq_l[t] = psq[t * 16 + lo];

    for (int s = 0; s < NSTRIPS; ++s) {
        const int row0 = blockIdx.x * BLK_PTS + s * 128 + wave * 16;

        v8f acc[4];
#pragma unroll
        for (int t = 0; t < 4; ++t) acc[t] = (v8f)0.0f;

        // K-loop: D = A(16x32 bf16) x B(32x16 bf16) + C, 8 steps of 32
#pragma unroll 2
        for (int k = 0; k < DIM; k += 32) {
            // A fragment: row = row0+lo, 16 contiguous f32 at k+hi*16,
            // converted in-register (v_cvt_pk_bf16_f32)
            const float* ap = x + (size_t)(row0 + lo) * DIM + k + hi * 16;
            v4f a0 = *(const v4f*)(ap + 0);
            v4f a1 = *(const v4f*)(ap + 4);
            v4f a2 = *(const v4f*)(ap + 8);
            v4f a3 = *(const v4f*)(ap + 12);
            v16bf A;
#pragma unroll
            for (int i = 0; i < 4; ++i) {
                A[i]      = (__bf16)a0[i];
                A[4 + i]  = (__bf16)a1[i];
                A[8 + i]  = (__bf16)a2[i];
                A[12 + i] = (__bf16)a3[i];
            }
#pragma unroll
            for (int t = 0; t < 4; ++t) {
                // B fragment: 16 contiguous bf16 of protos_bf16[n = t*16+lo]
                // starting at k + hi*16 -> one aligned 32B load, no converts
                const __bf16* bp = pbf + (size_t)(t * 16 + lo) * DIM + k + hi * 16;
                v16bf B = *(const v16bf*)bp;
                acc[t] = __builtin_amdgcn_wmma_f32_16x16x32_bf16(
                    false, A, false, B, (short)0, acc[t], false, false);
            }
        }

        // scatter d2 = p_sq - 2*dot (x_sq constant per row -> irrelevant to argmin)
#pragma unroll
        for (int t = 0; t < 4; ++t)
#pragma unroll
            for (int v = 0; v < 8; ++v) {
                const int m = v + 8 * hi;           // C/D layout: M = vgpr + 8*(lane>>4)
                d2s[wave][m * 64 + t * 16 + lo] = psq_l[t] - 2.0f * acc[t][v];
            }

        // per-row argmin over 64 clusters (lanes 0..15, one row each; same-wave
        // LDS ops are in-order so no barrier needed)
        int bestc = 0;
        if (lane < 16) {
            float bd = d2s[wave][lane * 64];
            for (int n = 1; n < 64; ++n) {
                const float d = d2s[wave][lane * 64 + n];
                if (d < bd) { bd = d; bestc = n; }
            }
            idx[row0 + lane] = bestc;
        }

        // segment-sum: broadcast each row's cluster, all 32 lanes accumulate
        // that row's 256 features into LDS with ds_add_f32 atomics
#pragma unroll
        for (int r = 0; r < 16; ++r) {
            const int c = __builtin_amdgcn_readlane(bestc, r);
            const float* xr = x + (size_t)(row0 + r) * DIM;
#pragma unroll
            for (int j = 0; j < 8; ++j)
                atomicAdd(&lsum[c * DIM + lane + 32 * j], xr[lane + 32 * j]);
            if (lane == 0) atomicAdd(&lcnt[c], 1.0f);
        }
    }

    __syncthreads();
    for (int i = tid; i < KC * DIM; i += 256) atomicAdd(&sums[i], lsum[i]);
    if (tid < KC) atomicAdd(&counts[tid], lcnt[tid]);
}

// ---------------------------------------------------------------------------
// update: protos_new = sums/(counts+eps); variation; done flag; p_sq; bf16
// mirror; zero accumulators.  1 block x 256 threads, 64 elements each.
// ---------------------------------------------------------------------------
__global__ void kmeans_update(float* __restrict__ protos, __bf16* __restrict__ pbf,
                              float* __restrict__ psq,
                              float* __restrict__ sums, float* __restrict__ counts,
                              int* __restrict__ done) {
    if (*done) return;  // where(done, protos_old, ...): keep everything

    __shared__ float red[256];
    __shared__ float psq_s[KC];
    const int tid = threadIdx.x;
    if (tid < KC) psq_s[tid] = 0.0f;
    __syncthreads();

    const int c   = tid >> 2;               // 4 threads per cluster row
    const float cnt = counts[c] + 1e-6f;
    const int base  = tid * 64;
    float lv = 0.0f, ps = 0.0f;
    for (int j = 0; j < 64; ++j) {
        const float pn = sums[base + j] / cnt;
        const float po = protos[base + j];
        const float dd = pn - po;
        lv += dd * dd;
        ps += pn * pn;
        protos[base + j] = pn;
        pbf[base + j]    = (__bf16)pn;
        sums[base + j]   = 0.0f;            // ready for next iteration
    }
    atomicAdd(&psq_s[c], ps);
    red[tid] = lv;
    __syncthreads();
    for (int s = 128; s > 0; s >>= 1) {
        if (tid < s) red[tid] += red[tid + s];
        __syncthreads();
    }
    if (tid == 0 && red[0] * (1.0f / (KC * DIM)) < 1e-4f) *done = 1;
    __syncthreads();
    if (tid < KC) { psq[tid] = psq_s[tid]; counts[tid] = 0.0f; }
}

// ---------------------------------------------------------------------------
// finalize: d_out = [protos (64*256 f32), idx (131072, as f32)]
// ---------------------------------------------------------------------------
__global__ void kmeans_finalize(const float* __restrict__ protos,
                                const int* __restrict__ idx,
                                float* __restrict__ out) {
    const int i = blockIdx.x * 256 + threadIdx.x;   // grid covers 131072
    if (i < KC * DIM) out[i] = protos[i];
    out[KC * DIM + i] = (float)idx[i];
}

extern "C" void kernel_launch(void* const* d_in, const int* in_sizes, int n_in,
                              void* d_out, int out_size, void* d_ws, size_t ws_size,
                              hipStream_t stream) {
    (void)in_sizes; (void)n_in; (void)out_size; (void)ws_size;
    const float* x = (const float*)d_in[0];
    char* ws = (char*)d_ws;

    float*  protos = (float*) (ws);                 //  64*256*4 = 65536 B
    float*  sums   = (float*) (ws + 65536);         //  65536 B
    float*  psq    = (float*) (ws + 131072);        //  256 B
    float*  counts = (float*) (ws + 131328);        //  256 B
    int*    done   = (int*)   (ws + 131584);        //  4 B (padded)
    __bf16* pbf    = (__bf16*)(ws + 131840);        //  64*256*2 = 32768 B
    int*    idx    = (int*)   (ws + 164864);        //  131072*4 B

    kmeans_init<<<KC, 256, 0, stream>>>(x, protos, pbf, psq, sums, counts, done);
    for (int it = 0; it < ITERS; ++it) {
        kmeans_assign<<<NBLOCKS, 256, 0, stream>>>(x, pbf, psq, idx, sums,
                                                   counts, done);
        kmeans_update<<<1, 256, 0, stream>>>(protos, pbf, psq, sums, counts, done);
    }
    kmeans_finalize<<<NPTS / 256, 256, 0, stream>>>(protos, idx, (float*)d_out);
}
